// MultiHeadAttention_31696858644489
// MI455X (gfx1250) — matrix operand order repaired and checked by reading, not compiled, not run on hardware
//
#include <hip/hip_runtime.h>
#include <hip/hip_bf16.h>

// ---------- types for gfx1250 WMMA ----------
typedef __bf16 bf16_t;
typedef __attribute__((ext_vector_type(8)))  bf16_t bf16x8;
typedef __attribute__((ext_vector_type(16))) bf16_t bf16x16;
typedef __attribute__((ext_vector_type(8)))  float  floatx8;

union Frag { bf16x16 v; bf16x8 h[2]; };

static __device__ __forceinline__ unsigned short f2bfu(float f) {
    unsigned u = __builtin_bit_cast(unsigned, f);
    u += 0x7fffu + ((u >> 16) & 1u);          // round-to-nearest-even
    return (unsigned short)(u >> 16);
}
static __device__ __forceinline__ unsigned pk2(float a, float b) {
    return (unsigned)f2bfu(a) | ((unsigned)f2bfu(b) << 16);
}

// ---------- gfx1250 async global->LDS copy (ASYNCcnt path) ----------
#ifndef __has_builtin
#define __has_builtin(x) 0
#endif
#if __has_builtin(__builtin_amdgcn_global_load_async_to_lds_b128)
#define HAVE_ASYNC_LDS 1
#else
#define HAVE_ASYNC_LDS 0
#endif

// builtin signature (from compiler diagnostic):
//   void (...)(v4i addrspace(1)* gsrc, v4i addrspace(3)* ldsdst, imm int offset, imm int cpol)
typedef int v4i __attribute__((vector_size(16)));
typedef __attribute__((address_space(1))) v4i* gv4i_p;
typedef __attribute__((address_space(3))) v4i* lv4i_p;

static __device__ __forceinline__ void wait_async0() {
#if __has_builtin(__builtin_amdgcn_s_wait_asynccnt)
    __builtin_amdgcn_s_wait_asynccnt(0);
#else
    asm volatile("s_wait_asynccnt 0" ::: "memory");
#endif
}

#if HAVE_ASYNC_LDS
static __device__ __forceinline__ void async_cp16(const unsigned short* g, unsigned short* l) {
    __builtin_amdgcn_global_load_async_to_lds_b128(
        (gv4i_p)(unsigned long long)(size_t)(const void*)(g),
        (lv4i_p)(unsigned)(size_t)(const void*)(l),
        0, 0);
}
#endif

// =====================================================================
// GEMM: C[M=8192, N=1024] = A[M,1024] x W[1024,1024] + bias
// AMODE 0: A fp32 (convert to bf16), 1: A bf16 (ushort, async copy)
// OMODE 0: fp32 plain store, 1: bf16 head-split store [B,H,S,64] * oscale
// Block: 256 thr (8 waves), 128x128 tile; wave tile 32x64 (2x4 wmma)
// =====================================================================
template<int AMODE, int OMODE>
__global__ __launch_bounds__(256) void gemm128_wmma(
    const float* __restrict__ A32, const unsigned short* __restrict__ A16,
    const float* __restrict__ W,   const float* __restrict__ bias,
    float* __restrict__ out32,     unsigned short* __restrict__ out16,
    float oscale)
{
    constexpr int K = 1024, N = 1024;
    __shared__ unsigned short As[128 * 40];   // [m][k], stride 40 halves (80B)
    __shared__ unsigned short Bs[128 * 40];   // [n][k] (W transposed), stride 40

    const int tid  = threadIdx.x;
    const int lane = tid & 31;
    const int w    = tid >> 5;
    const int wm   = w & 3;          // 4 waves along M
    const int wn   = w >> 2;         // 2 waves along N
    const int m0   = blockIdx.y * 128;
    const int n0   = blockIdx.x * 128;
    const int lh   = lane & 15;
    const int half = lane >> 4;

    floatx8 acc[2][4];
    const floatx8 z8 = {0.f,0.f,0.f,0.f,0.f,0.f,0.f,0.f};
    #pragma unroll
    for (int i = 0; i < 2; ++i)
        #pragma unroll
        for (int j = 0; j < 4; ++j) acc[i][j] = z8;

    for (int kk = 0; kk < K; kk += 32) {
        __syncthreads();
        // ---- stage A 128x32 ----
        if (AMODE == 0) {
            #pragma unroll
            for (int i = 0; i < 4; ++i) {
                int fi  = i * 256 + tid;
                int row = fi >> 3, c4 = (fi & 7) << 2;
                const float4 a = *(const float4*)(A32 + (size_t)(m0 + row) * K + kk + c4);
                uint2 d; d.x = pk2(a.x, a.y); d.y = pk2(a.z, a.w);
                *(uint2*)(As + row * 40 + c4) = d;
            }
        } else {
#if HAVE_ASYNC_LDS
            // async DMA: 512 x 16B chunks, 2 per thread, no VGPR round-trip
            #pragma unroll
            for (int i = 0; i < 2; ++i) {
                int ci  = i * 256 + tid;
                int row = ci >> 2, c8 = (ci & 3) << 3;
                async_cp16(A16 + (size_t)(m0 + row) * K + kk + c8, As + row * 40 + c8);
            }
#else
            #pragma unroll
            for (int i = 0; i < 4; ++i) {
                int fi  = i * 256 + tid;
                int row = fi >> 3, c4 = (fi & 7) << 2;
                *(uint2*)(As + row * 40 + c4) =
                    *(const uint2*)(A16 + (size_t)(m0 + row) * K + kk + c4);
            }
#endif
        }
        // ---- stage W 32x128 transposed into Bs[n][k] (fp32->bf16) ----
        #pragma unroll
        for (int i = 0; i < 4; ++i) {
            int fi = i * 256 + tid;
            int kr = fi >> 5, n4 = (fi & 31) << 2;
            const float4 wv = *(const float4*)(W + (size_t)(kk + kr) * N + n0 + n4);
            Bs[(n4 + 0) * 40 + kr] = f2bfu(wv.x);
            Bs[(n4 + 1) * 40 + kr] = f2bfu(wv.y);
            Bs[(n4 + 2) * 40 + kr] = f2bfu(wv.z);
            Bs[(n4 + 3) * 40 + kr] = f2bfu(wv.w);
        }
        if (kk + 32 < K) {   // prefetch next tiles (global_prefetch_b8)
            if (AMODE == 0)
                __builtin_prefetch(A32 + (size_t)(m0 + (tid >> 1)) * K + kk + 32 + (tid & 1) * 16, 0, 1);
            __builtin_prefetch(W + (size_t)(kk + 32 + (tid >> 3)) * N + n0 + (tid & 7) * 16, 0, 1);
        }
#if HAVE_ASYNC_LDS
        if (AMODE == 1) wait_async0();
#endif
        __syncthreads();

        // ---- fragments: contiguous k = 16*half + s, same order for A and B ----
        Frag a[2], b[4];
        #pragma unroll
        for (int mi = 0; mi < 2; ++mi) {
            const unsigned short* p = As + (32 * wm + 16 * mi + lh) * 40 + half * 16;
            a[mi].h[0] = *(const bf16x8*)(p);        // K contiguous 16
            a[mi].h[1] = *(const bf16x8*)(p + 8);
        }
        #pragma unroll
        for (int ni = 0; ni < 4; ++ni) {
            const unsigned short* p = Bs + (64 * wn + 16 * ni + lh) * 40 + half * 16;
            b[ni].h[0] = *(const bf16x8*)(p);        // K contiguous 16
            b[ni].h[1] = *(const bf16x8*)(p + 8);
        }
        #pragma unroll
        for (int mi = 0; mi < 2; ++mi)
            #pragma unroll
            for (int ni = 0; ni < 4; ++ni)
                acc[mi][ni] = __builtin_amdgcn_wmma_f32_16x16x32_bf16(
                    false, a[mi].v, false, b[ni].v, (short)0, acc[mi][ni], false, false);
    }

    // ---- epilogue: C/D layout row = vgpr + 8*half, col = lane&15 ----
    #pragma unroll
    for (int mi = 0; mi < 2; ++mi)
        #pragma unroll
        for (int ni = 0; ni < 4; ++ni)
            #pragma unroll
            for (int r = 0; r < 8; ++r) {
                int row = m0 + 32 * wm + 16 * mi + r + 8 * half;
                int col = n0 + 64 * wn + 16 * ni + lh;
                float val = acc[mi][ni][r] + bias[col];
                if (OMODE == 0) {
                    out32[(size_t)row * N + col] = val;
                } else {
                    int bb = row >> 11, s = row & 2047;   // S = 2048
                    int hh = col >> 6,  d = col & 63;     // d_k = 64
                    out16[(((size_t)bb * 16 + hh) * 2048 + s) * 64 + d] = f2bfu(val * oscale);
                }
            }
}

// =====================================================================
// Flash attention: per (b, h, 64-query tile). 128 thr = 4 waves,
// each wave: 16 queries x 64 keys scores, 16 x 64 d_k output.
// Q/K/V are bf16 [B,H,S,64] in workspace; Q pre-scaled by 1/sqrt(d_k).
// =====================================================================
__global__ __launch_bounds__(128) void attn_wmma(
    const unsigned short* __restrict__ Qb,
    const unsigned short* __restrict__ Kb,
    const unsigned short* __restrict__ Vb,
    unsigned short* __restrict__ Ob)
{
    __shared__ unsigned short Qs [64 * 72];   // [q][d]
    __shared__ unsigned short Ks [64 * 72];   // [key][d]  (== K^T B-frag layout)
    __shared__ unsigned short VTs[64 * 72];   // [d][key]  (V transposed)
    __shared__ unsigned short Ps [64 * 72];   // [q][key]  probabilities

    const int tid  = threadIdx.x;
    const int lane = tid & 31;
    const int wq   = tid >> 5;     // 0..3, 16 queries per wave
    const int lh   = lane & 15;
    const int half = lane >> 4;

    const int qt = blockIdx.x, h = blockIdx.y, b = blockIdx.z;
    const int q0 = qt * 64;
    const size_t head_base = ((size_t)b * 16 + h) * 2048 * 64;

    // stage Q tile once (pure bf16 copy -> async DMA when available)
#if HAVE_ASYNC_LDS
    #pragma unroll
    for (int i = 0; i < 4; ++i) {
        int ci  = i * 128 + tid;                 // 512 x 16B chunks
        int row = ci >> 3, c8 = (ci & 7) << 3;
        async_cp16(Qb + head_base + (size_t)(q0 + row) * 64 + c8, Qs + row * 72 + c8);
    }
#else
    #pragma unroll
    for (int i = 0; i < 8; ++i) {
        int fi = i * 128 + tid;
        int row = fi >> 4, c4 = (fi & 15) << 2;
        *(uint2*)(Qs + row * 72 + c4) =
            *(const uint2*)(Qb + head_base + (size_t)(q0 + row) * 64 + c4);
    }
#endif

    floatx8 accO[4];
    const floatx8 z8 = {0.f,0.f,0.f,0.f,0.f,0.f,0.f,0.f};
    float m_run[8], l_run[8];
    #pragma unroll
    for (int ni = 0; ni < 4; ++ni) accO[ni] = z8;
    #pragma unroll
    for (int r = 0; r < 8; ++r) { m_run[r] = -3.0e38f; l_run[r] = 0.0f; }

    for (int kt = 0; kt < 32; ++kt) {
        __syncthreads();
        // K tile: pure copy -> async DMA; V tile: transposed manually
#if HAVE_ASYNC_LDS
        #pragma unroll
        for (int i = 0; i < 4; ++i) {
            int ci  = i * 128 + tid;
            int row = ci >> 3, c8 = (ci & 7) << 3;
            async_cp16(Kb + head_base + (size_t)(kt * 64 + row) * 64 + c8, Ks + row * 72 + c8);
        }
#endif
        #pragma unroll
        for (int i = 0; i < 8; ++i) {
            int fi = i * 128 + tid;
            int row = fi >> 4, c4 = (fi & 15) << 2;
#if !HAVE_ASYNC_LDS
            *(uint2*)(Ks + row * 72 + c4) =
                *(const uint2*)(Kb + head_base + (size_t)(kt * 64 + row) * 64 + c4);
#endif
            uint2 vv = *(const uint2*)(Vb + head_base + (size_t)(kt * 64 + row) * 64 + c4);
            VTs[(c4 + 0) * 72 + row] = (unsigned short)(vv.x & 0xffffu);
            VTs[(c4 + 1) * 72 + row] = (unsigned short)(vv.x >> 16);
            VTs[(c4 + 2) * 72 + row] = (unsigned short)(vv.y & 0xffffu);
            VTs[(c4 + 3) * 72 + row] = (unsigned short)(vv.y >> 16);
        }
#if HAVE_ASYNC_LDS
        wait_async0();
#endif
        __syncthreads();

        // ---- S = Q K^T (already scaled; d_k = 64 -> two K=32 chunks) ----
        floatx8 sa[4];
        #pragma unroll
        for (int ni = 0; ni < 4; ++ni) sa[ni] = z8;
        #pragma unroll
        for (int c = 0; c < 2; ++c) {
            Frag aq;
            const unsigned short* ap = Qs + (16 * wq + lh) * 72 + c * 32 + half * 16;
            aq.h[0] = *(const bf16x8*)(ap);
            aq.h[1] = *(const bf16x8*)(ap + 8);
            #pragma unroll
            for (int ni = 0; ni < 4; ++ni) {
                Frag bk;
                const unsigned short* bp = Ks + (16 * ni + lh) * 72 + c * 32 + half * 16;
                bk.h[0] = *(const bf16x8*)(bp);
                bk.h[1] = *(const bf16x8*)(bp + 8);
                sa[ni] = __builtin_amdgcn_wmma_f32_16x16x32_bf16(
                    false, aq.v, false, bk.v, (short)0, sa[ni], false, false);
            }
        }

        // ---- online softmax (rows striped across 16 lanes) ----
        float mx[8];
        #pragma unroll
        for (int r = 0; r < 8; ++r)
            mx[r] = fmaxf(fmaxf(sa[0][r], sa[1][r]), fmaxf(sa[2][r], sa[3][r]));
        #pragma unroll
        for (int off = 1; off < 16; off <<= 1)
            #pragma unroll
            for (int r = 0; r < 8; ++r)
                mx[r] = fmaxf(mx[r], __shfl_xor(mx[r], off, 16));

        float fr[8], rs[8];
        #pragma unroll
        for (int r = 0; r < 8; ++r) {
            float mn = fmaxf(m_run[r], mx[r]);
            fr[r] = __expf(m_run[r] - mn);
            m_run[r] = mn;
            rs[r] = 0.0f;
        }
        #pragma unroll
        for (int ni = 0; ni < 4; ++ni)
            #pragma unroll
            for (int r = 0; r < 8; ++r) {
                float p = __expf(sa[ni][r] - m_run[r]);
                sa[ni][r] = p;
                rs[r] += p;
            }
        #pragma unroll
        for (int off = 1; off < 16; off <<= 1)
            #pragma unroll
            for (int r = 0; r < 8; ++r)
                rs[r] += __shfl_xor(rs[r], off, 16);
        #pragma unroll
        for (int r = 0; r < 8; ++r) l_run[r] = l_run[r] * fr[r] + rs[r];
        #pragma unroll
        for (int ni = 0; ni < 4; ++ni)
            #pragma unroll
            for (int r = 0; r < 8; ++r) accO[ni][r] *= fr[r];

        // write P (wave-private rows; same-wave LDS ops are ordered)
        #pragma unroll
        for (int ni = 0; ni < 4; ++ni)
            #pragma unroll
            for (int r = 0; r < 8; ++r)
                Ps[(16 * wq + r + 8 * half) * 72 + 16 * ni + lh] = f2bfu(sa[ni][r]);

        // ---- O += P V  (keys = K dim, two K=32 chunks) ----
        #pragma unroll
        for (int c = 0; c < 2; ++c) {
            Frag ap;
            const unsigned short* pp = Ps + (16 * wq + lh) * 72 + c * 32 + half * 16;
            ap.h[0] = *(const bf16x8*)(pp);
            ap.h[1] = *(const bf16x8*)(pp + 8);
            #pragma unroll
            for (int ni = 0; ni < 4; ++ni) {
                Frag bv;
                const unsigned short* bp = VTs + (16 * ni + lh) * 72 + c * 32 + half * 16;
                bv.h[0] = *(const bf16x8*)(bp);
                bv.h[1] = *(const bf16x8*)(bp + 8);
                accO[ni] = __builtin_amdgcn_wmma_f32_16x16x32_bf16(
                    false, ap.v, false, bv.v, (short)0, accO[ni], false, false);
            }
        }
    }

    // normalize and store merged-head layout [B, S, H*64] bf16
    #pragma unroll
    for (int ni = 0; ni < 4; ++ni)
        #pragma unroll
        for (int r = 0; r < 8; ++r) {
            int qrow = q0 + 16 * wq + r + 8 * half;
            int d    = 16 * ni + lh;
            float v  = accO[ni][r] / l_run[r];
            Ob[((size_t)(b * 2048 + qrow)) * 1024 + h * 64 + d] = f2bfu(v);
        }
}

// =====================================================================
extern "C" void kernel_launch(void* const* d_in, const int* in_sizes, int n_in,
                              void* d_out, int out_size, void* d_ws, size_t ws_size,
                              hipStream_t stream) {
    (void)in_sizes; (void)n_in; (void)out_size; (void)ws_size;
    const float* q  = (const float*)d_in[0];
    const float* k  = (const float*)d_in[1];
    const float* v  = (const float*)d_in[2];
    const float* Wq = (const float*)d_in[3];
    const float* bq = (const float*)d_in[4];
    const float* Wk = (const float*)d_in[5];
    const float* bk = (const float*)d_in[6];
    const float* Wv = (const float*)d_in[7];
    const float* bv = (const float*)d_in[8];
    const float* Wo = (const float*)d_in[9];
    const float* bo = (const float*)d_in[10];
    float* out = (float*)d_out;

    const size_t HEADBUF = (size_t)4 * 16 * 2048 * 64;   // bf16 elements
    unsigned short* Qb = (unsigned short*)d_ws;
    unsigned short* Kb = Qb + HEADBUF;
    unsigned short* Vb = Kb + HEADBUF;
    unsigned short* Ob = Vb + HEADBUF;

    dim3 gg(8, 64), bb(256);
    // Q is pre-scaled by 1/sqrt(d_k) = 0.125 so attention needs no scaling pass
    gemm128_wmma<0, 1><<<gg, bb, 0, stream>>>(q, nullptr, Wq, bq, nullptr, Qb, 0.125f);
    gemm128_wmma<0, 1><<<gg, bb, 0, stream>>>(k, nullptr, Wk, bk, nullptr, Kb, 1.0f);
    gemm128_wmma<0, 1><<<gg, bb, 0, stream>>>(v, nullptr, Wv, bv, nullptr, Vb, 1.0f);
    attn_wmma<<<dim3(32, 16, 4), 128, 0, stream>>>(Qb, Kb, Vb, Ob);
    gemm128_wmma<1, 0><<<gg, bb, 0, stream>>>(nullptr, Ob, Wo, bo, out, nullptr, 1.0f);
}
